// EncoderMambaBlock_28003186769941
// MI455X (gfx1250) — matrix operand order repaired and checked by reading, not compiled, not run on hardware
//
#include <hip/hip_runtime.h>
#include <hip/hip_bf16.h>
#include <math.h>

#define DIM 1024
#define DT_RANK 64
#define D_STATE 16
#define BATCH 2
#define SEQ 2048
#define MROWS (BATCH * SEQ)
#define EPS 1e-5f

typedef __attribute__((ext_vector_type(16))) _Float16 v16h;
typedef __attribute__((ext_vector_type(8)))  float    v8f;
typedef __attribute__((ext_vector_type(4)))  unsigned int u32x4;
typedef __attribute__((ext_vector_type(4)))  int       i32x4;
typedef __attribute__((ext_vector_type(8)))  int       i32x8;

__device__ __forceinline__ float softplusf(float x) {
    return x > 20.f ? x : log1pf(__expf(x));
}
__device__ __forceinline__ float sigmoidf_(float x) {
    return 1.f / (1.f + __expf(-x));
}

// ---------------------------------------------------------------------------
// TDM: issue one 2D tensor_load_to_lds (global -> LDS), 16-bit elements.
// Descriptor per CDNA5 ISA ch.8: group0 = {count, lds_addr, global_addr, type=2},
// group1 = {data_size=2B, tensor_dim0/1, tile_dim0/1, tensor_dim0_stride}.
// All dims/strides in 2-byte units. Wave-uniform (call from one wave only).
// ---------------------------------------------------------------------------
__device__ __forceinline__ void tdm_load_2d(unsigned lds_addr, const void* gaddr,
                                            unsigned tensor_d0, unsigned tensor_d1,
                                            unsigned tile_d0, unsigned tile_d1,
                                            unsigned stride0) {
    unsigned long long ga = (unsigned long long)(size_t)gaddr;
    unsigned ga_lo = __builtin_amdgcn_readfirstlane((unsigned)ga);
    unsigned ga_hi = __builtin_amdgcn_readfirstlane((unsigned)(ga >> 32));
    lds_addr = __builtin_amdgcn_readfirstlane(lds_addr);

    u32x4 g0 = {
        1u,                                        // [1:0] count=1 (valid user D#)
        lds_addr,                                  // [63:32] LDS byte address
        ga_lo,                                     // [95:64] global addr low
        (ga_hi & 0x01FFFFFFu) | (2u << 30)         // [120:96] addr hi, [127:126] type=2
    };
    i32x8 g1 = {
        (int)(1u << 16),                                          // data_size=1 (2 bytes)
        (int)((tensor_d0 & 0xFFFFu) << 16),                       // tensor_dim0 [79:48] lo
        (int)((tensor_d0 >> 16) | ((tensor_d1 & 0xFFFFu) << 16)), // dim0 hi | tensor_dim1 lo
        (int)((tensor_d1 >> 16) | (tile_d0 << 16)),               // dim1 hi | tile_dim0
        (int)(tile_d1 & 0xFFFFu),                                 // tile_dim1 (tile_dim2=0)
        (int)stride0,                                             // tensor_dim0_stride lo32
        0, 0                                                      // stride hi / dim1_stride
    };
    i32x4 z4 = { 0, 0, 0, 0 };
#if defined(__clang_major__) && (__clang_major__ >= 23)
    i32x8 z8 = { 0, 0, 0, 0, 0, 0, 0, 0 };
    __builtin_amdgcn_tensor_load_to_lds(g0, g1, z4, z4, z8, 0);
#else
    __builtin_amdgcn_tensor_load_to_lds(g0, g1, z4, z4, 0);
#endif
}

// ---------------------------------------------------------------------------
// Weight convert (f32 -> f16) + transpose: W[N][K] row-major -> Wt[K][N]
// ---------------------------------------------------------------------------
__global__ void wcvt_kernel(const float* __restrict__ W, _Float16* __restrict__ Wt,
                            int N, int K) {
    int idx = blockIdx.x * blockDim.x + threadIdx.x;
    if (idx >= N * K) return;
    int n = idx / K, k = idx % K;
    Wt[(size_t)k * N + n] = (_Float16)W[idx];
}

// ---------------------------------------------------------------------------
// LayerNorm over last dim (1024); writes f16 activations for the WMMA GEMMs
// ---------------------------------------------------------------------------
__global__ void ln_kernel(const float* __restrict__ x, const float* __restrict__ w,
                          const float* __restrict__ b, _Float16* __restrict__ xn) {
    int row = blockIdx.x;
    const float* xr = x + (size_t)row * DIM;
    float s = 0.f, ss = 0.f;
    for (int i = threadIdx.x; i < DIM; i += 256) {
        float v = xr[i];
        s += v; ss += v * v;
    }
    __shared__ float sb0[256];
    __shared__ float sb1[256];
    sb0[threadIdx.x] = s; sb1[threadIdx.x] = ss;
    __syncthreads();
    for (int off = 128; off > 0; off >>= 1) {
        if (threadIdx.x < off) {
            sb0[threadIdx.x] += sb0[threadIdx.x + off];
            sb1[threadIdx.x] += sb1[threadIdx.x + off];
        }
        __syncthreads();
    }
    float mu  = sb0[0] * (1.f / DIM);
    float var = sb1[0] * (1.f / DIM) - mu * mu;
    float inv = rsqrtf(var + EPS);
    for (int i = threadIdx.x; i < DIM; i += 256) {
        float v = (xr[i] - mu) * inv * w[i] + b[i];
        xn[(size_t)row * DIM + i] = (_Float16)v;
    }
}

union AFrag { v16h v; unsigned int u[8]; };
union DFrag { v8f  v; float f[8]; };

// ---------------------------------------------------------------------------
// TDM + LDS double-buffered WMMA GEMM:
//   C[M][N] (f32) = A[M][lda](f16, K cols) @ Wt[K][N](f16)
// Block = 256 thr (8 waves, 4 m-rows x 2 n-cols), block tile 128x128,
// K stepped by 32. Wave 0 issues two TDM DMAs per step into the next LDS
// buffer, waits TENSORcnt<=2 (in-order => current buffer complete), then the
// block barriers and all 8 waves run 8 WMMAs each from LDS fragments.
// EPI: 0 = +bias, 1 = +bias then softplus, 2 = raw
// ---------------------------------------------------------------------------
template <int EPI>
__global__ void __launch_bounds__(256)
gemm_tdm_kernel(const _Float16* __restrict__ A, int lda,
                const _Float16* __restrict__ Wt, const float* __restrict__ bias,
                float* __restrict__ C, _Float16* __restrict__ Ch,
                int M, int N, int K) {
    __shared__ _Float16 As[2][128 * 32];   // 8KB per buffer, row-major 128x32
    __shared__ _Float16 Bs[2][32 * 128];   // 8KB per buffer, row-major 32x128

    const int nblk = N / 128;
    const int row0 = (blockIdx.x / nblk) * 128;
    const int col0 = (blockIdx.x % nblk) * 128;
    const int tid    = threadIdx.x;
    const int waveId = tid >> 5;
    const int lane   = tid & 31;
    const int r  = lane & 15;
    const int hi = lane >> 4;
    const int mb = (waveId & 3) * 32;      // wave row offset in block tile
    const int nb = (waveId >> 2) * 64;     // wave col offset in block tile

    const int ksteps = K / 32;
    v8f acc[2][4] = {};

    if (waveId == 0) {
        tdm_load_2d((unsigned)(size_t)&As[0][0], A + (size_t)row0 * lda,
                    (unsigned)lda, (unsigned)M, 32u, 128u, (unsigned)lda);
        tdm_load_2d((unsigned)(size_t)&Bs[0][0], Wt + (size_t)col0,
                    (unsigned)N, (unsigned)K, 128u, 32u, (unsigned)N);
    }

    for (int ks = 0; ks < ksteps; ++ks) {
        const int buf = ks & 1;
        if (waveId == 0) {
            if (ks + 1 < ksteps) {
                const int k1 = (ks + 1) * 32;
                tdm_load_2d((unsigned)(size_t)&As[buf ^ 1][0],
                            A + (size_t)row0 * lda + k1,
                            (unsigned)lda, (unsigned)M, 32u, 128u, (unsigned)lda);
                tdm_load_2d((unsigned)(size_t)&Bs[buf ^ 1][0],
                            Wt + (size_t)k1 * N + col0,
                            (unsigned)N, (unsigned)K, 128u, 32u, (unsigned)N);
                __builtin_amdgcn_s_wait_tensorcnt((short)2);
            } else {
                __builtin_amdgcn_s_wait_tensorcnt((short)0);
            }
        }
        __syncthreads();

        AFrag af[2];
        const unsigned int* asu = (const unsigned int*)&As[buf][0];
#pragma unroll
        for (int mt = 0; mt < 2; ++mt) {
            const int rowl = (mb + mt * 16 + r) * 16;   // 16 dwords per 32-half row
#pragma unroll
            for (int p = 0; p < 4; ++p) af[mt].u[p]     = asu[rowl + 4 * hi + p];
#pragma unroll
            for (int p = 0; p < 4; ++p) af[mt].u[4 + p] = asu[rowl + 8 + 4 * hi + p];
        }
        const _Float16* bsl = &Bs[buf][nb + r];
#pragma unroll
        for (int nt = 0; nt < 4; ++nt) {
            v16h bf;
#pragma unroll
            for (int s = 0; s < 16; ++s)
                bf[s] = bsl[((s < 8) ? (8 * hi + s) : (16 + 8 * hi + s - 8)) * 128 + nt * 16];
#pragma unroll
            for (int mt = 0; mt < 2; ++mt)
                acc[mt][nt] = __builtin_amdgcn_wmma_f32_16x16x32_f16(
                    false, af[mt].v, false, bf, (short)0, acc[mt][nt], false, false);
        }
        __syncthreads();
    }

#pragma unroll
    for (int mt = 0; mt < 2; ++mt) {
#pragma unroll
        for (int nt = 0; nt < 4; ++nt) {
            const int col = col0 + nb + nt * 16 + r;
            float bv = (EPI != 2 && bias) ? bias[col] : 0.f;
            DFrag d; d.v = acc[mt][nt];
#pragma unroll
            for (int rr = 0; rr < 8; ++rr) {
                const int row = row0 + mb + mt * 16 + rr + 8 * hi;
                float val = d.f[rr] + bv;
                if (EPI == 1) val = softplusf(val);
                C[(size_t)row * N + col] = val;
                if (Ch) Ch[(size_t)row * N + col] = (_Float16)val;
            }
        }
    }
}

// ---------------------------------------------------------------------------
// Direct-global WMMA GEMM (used for the narrow N=96 deltaBC projection).
// Each wave computes a 16 x (16*NT) strip; B fragment is gathered from 16 halves strided by N.
// ---------------------------------------------------------------------------
template <int NT, int EPI>
__global__ void gemm_wmma_kernel(const _Float16* __restrict__ A, int lda,
                                 const _Float16* __restrict__ Wt,
                                 const float* __restrict__ bias,
                                 float* __restrict__ C, _Float16* __restrict__ Ch,
                                 int M, int N, int K) {
    int w = (blockIdx.x * blockDim.x + threadIdx.x) >> 5;
    int tiles_n = N / (16 * NT);
    int total = (M / 16) * tiles_n;
    if (w >= total) return;                 // wave-uniform: EXEC stays all-ones
    int col0 = (w % tiles_n) * (16 * NT);
    int row0 = (w / tiles_n) * 16;
    int lane = threadIdx.x & 31;
    int r  = lane & 15;
    int hi = lane >> 4;

    v8f acc[NT] = {};
    const _Float16* arow = A + (size_t)(row0 + r) * lda;
    for (int k0 = 0; k0 < K; k0 += 32) {
        AFrag af;
        const unsigned int* ap = (const unsigned int*)(arow + k0);
#pragma unroll
        for (int p = 0; p < 4; p++) af.u[p]     = ap[4 * hi + p];
#pragma unroll
        for (int p = 0; p < 4; p++) af.u[4 + p] = ap[8 + 4 * hi + p];
        const _Float16* brow = Wt + (size_t)k0 * N + col0 + r;
#pragma unroll
        for (int t = 0; t < NT; t++) {
            v16h bf;
#pragma unroll
            for (int s = 0; s < 16; s++)
                bf[s] = brow[(size_t)((s < 8) ? (8 * hi + s) : (16 + 8 * hi + s - 8)) * N + 16 * t];
            acc[t] = __builtin_amdgcn_wmma_f32_16x16x32_f16(
                false, af.v, false, bf, (short)0, acc[t], false, false);
        }
    }
#pragma unroll
    for (int t = 0; t < NT; t++) {
        int col  = col0 + 16 * t + (lane & 15);
        float bv = (EPI != 2 && bias) ? bias[col] : 0.f;
        DFrag d; d.v = acc[t];
#pragma unroll
        for (int rr = 0; rr < 8; rr++) {
            int row = row0 + rr + 8 * hi;
            float val = d.f[rr] + bv;
            if (EPI == 1) val = softplusf(val);
            C[(size_t)row * N + col] = val;
            if (Ch) Ch[(size_t)row * N + col] = (_Float16)val;
        }
    }
}

// ---------------------------------------------------------------------------
// Selective scan. One lane per (batch, channel d, state n): 32768 lanes,
// SEQ serial steps. 16-lane shuffle-xor tree reduces y = sum_n h*C.
// dbc rows are [delta_raw(64) | B(16) | C(16)] per (b,l).
// ---------------------------------------------------------------------------
__global__ void scan_kernel(const float* __restrict__ u, const float* __restrict__ delta,
                            const float* __restrict__ dbc, const float* __restrict__ A_log,
                            const float* __restrict__ Dv, float* __restrict__ y) {
    int gid = blockIdx.x * blockDim.x + threadIdx.x;  // 0 .. 32767
    int ch  = gid >> 4;                                // (b,d) channel
    int n   = gid & 15;
    int b = ch / DIM, d = ch % DIM;
    float a  = -__expf(A_log[d * D_STATE + n]);
    float Dd = Dv[d];
    float h  = 0.f;
    const size_t base_u  = (size_t)b * SEQ * DIM + d;
    const size_t base_bc = (size_t)b * SEQ * 96;
    for (int l = 0; l < SEQ; ++l) {
        float dlt = delta[base_u + (size_t)l * DIM];
        float uu  = u[base_u + (size_t)l * DIM];
        const float* bc = dbc + base_bc + (size_t)l * 96;
        float Bv = bc[DT_RANK + n];
        float Cv = bc[DT_RANK + D_STATE + n];
        h = __expf(dlt * a) * h + dlt * uu * Bv;
        float yv = h * Cv;
        yv += __shfl_xor(yv, 8, 32);
        yv += __shfl_xor(yv, 4, 32);
        yv += __shfl_xor(yv, 2, 32);
        yv += __shfl_xor(yv, 1, 32);
        if (n == 0) y[base_u + (size_t)l * DIM] = yv + uu * Dd;
    }
}

// ---------------------------------------------------------------------------
// out = (y1 + y2) * silu(xp) + skip
// ---------------------------------------------------------------------------
__global__ void combine_kernel(const float* __restrict__ y1, const float* __restrict__ y2,
                               const float* __restrict__ xp, const float* __restrict__ skip,
                               float* __restrict__ out, int n) {
    int i = blockIdx.x * blockDim.x + threadIdx.x;
    if (i >= n) return;
    float xv = xp[i];
    float z  = xv * sigmoidf_(xv);
    out[i] = (y1[i] + y2[i]) * z + skip[i];
}

// ---------------------------------------------------------------------------
extern "C" void kernel_launch(void* const* d_in, const int* in_sizes, int n_in,
                              void* d_out, int out_size, void* d_ws, size_t ws_size,
                              hipStream_t stream) {
    const float* x        = (const float*)d_in[0];
    const float* ln_w     = (const float*)d_in[1];
    const float* ln_b     = (const float*)d_in[2];
    const float* proj_W   = (const float*)d_in[3];
    const float* proj_b   = (const float*)d_in[4];
    const float* fconv_W  = (const float*)d_in[5];
    const float* fconv_b  = (const float*)d_in[6];
    const float* bconv_W  = (const float*)d_in[7];
    const float* bconv_b  = (const float*)d_in[8];
    const float* dbc_W    = (const float*)d_in[9];
    const float* dt_W     = (const float*)d_in[10];
    const float* dt_b     = (const float*)d_in[11];
    const float* A_log    = (const float*)d_in[12];
    const float* Dvec     = (const float*)d_in[13];
    float* out = (float*)d_out;

    char* ws = (char*)d_ws;
    size_t off = 0;
    auto alloc = [&](size_t bytes) -> void* {
        void* p = ws + off;
        off += (bytes + 255) & ~(size_t)255;
        return p;
    };
    _Float16* xn_h    = (_Float16*)alloc((size_t)MROWS * DIM * 2);
    _Float16* projWt  = (_Float16*)alloc((size_t)DIM * DIM * 2);
    _Float16* fconvWt = (_Float16*)alloc((size_t)DIM * DIM * 2);
    _Float16* bconvWt = (_Float16*)alloc((size_t)DIM * DIM * 2);
    _Float16* dbcWt   = (_Float16*)alloc((size_t)DIM * 96 * 2);   // 1024 x 96
    _Float16* dtWt    = (_Float16*)alloc((size_t)DT_RANK * DIM * 2);
    float*    xp_f    = (float*)alloc((size_t)MROWS * DIM * 4);
    _Float16* xp_h    = (_Float16*)alloc((size_t)MROWS * DIM * 2);
    float*    u_f     = (float*)alloc((size_t)MROWS * DIM * 4);
    _Float16* u_h     = (_Float16*)alloc((size_t)MROWS * DIM * 2);
    float*    dbc_f   = (float*)alloc((size_t)MROWS * 96 * 4);
    _Float16* dbc_h   = (_Float16*)alloc((size_t)MROWS * 96 * 2);
    float*    delta_f = (float*)alloc((size_t)MROWS * DIM * 4);
    float*    y1      = (float*)alloc((size_t)MROWS * DIM * 4);
    float*    y2      = (float*)alloc((size_t)MROWS * DIM * 4);

    // --- weight convert + transpose (once per launch) ---
    {
        int tot = DIM * DIM, blk = (tot + 255) / 256;
        wcvt_kernel<<<blk, 256, 0, stream>>>(proj_W, projWt, DIM, DIM);
        wcvt_kernel<<<blk, 256, 0, stream>>>(fconv_W, fconvWt, DIM, DIM);
        wcvt_kernel<<<blk, 256, 0, stream>>>(bconv_W, bconvWt, DIM, DIM);
        int tot2 = 96 * DIM;
        wcvt_kernel<<<(tot2 + 255) / 256, 256, 0, stream>>>(dbc_W, dbcWt, 96, DIM);
        int tot3 = DIM * DT_RANK;
        wcvt_kernel<<<(tot3 + 255) / 256, 256, 0, stream>>>(dt_W, dtWt, DIM, DT_RANK);
    }

    // --- layernorm ---
    ln_kernel<<<MROWS, 256, 0, stream>>>(x, ln_w, ln_b, xn_h);

    const int big_blocks = (MROWS / 128) * (DIM / 128);   // 32 * 8 = 256

    // --- xp = xn @ proj_W.T + proj_b ---
    gemm_tdm_kernel<0><<<big_blocks, 256, 0, stream>>>(
        xn_h, DIM, projWt, proj_b, xp_f, xp_h, MROWS, DIM, DIM);

    const _Float16* brWt[2] = { fconvWt, bconvWt };
    const float*    brB[2]  = { fconv_b, bconv_b };
    float*          brY[2]  = { y1, y2 };

    for (int br = 0; br < 2; br++) {
        // u = softplus(xp @ W.T + b)
        gemm_tdm_kernel<1><<<big_blocks, 256, 0, stream>>>(
            xp_h, DIM, brWt[br], brB[br], u_f, u_h, MROWS, DIM, DIM);
        // dbc = u @ deltaBC_W.T   (N = 96, no bias) — direct-global WMMA path
        {
            int waves = (MROWS / 16) * (96 / 32);
            gemm_wmma_kernel<2, 2><<<(waves * 32 + 255) / 256, 256, 0, stream>>>(
                u_h, DIM, dbcWt, nullptr, dbc_f, dbc_h, MROWS, 96, DIM);
        }
        // delta = softplus(delta_raw @ dt_W.T + dt_b)  (A = dbc_h cols 0..63, lda=96)
        gemm_tdm_kernel<1><<<big_blocks, 256, 0, stream>>>(
            dbc_h, 96, dtWt, dt_b, delta_f, (_Float16*)nullptr, MROWS, DIM, DT_RANK);
        // selective scan
        scan_kernel<<<(BATCH * DIM * D_STATE) / 256, 256, 0, stream>>>(
            u_f, delta_f, dbc_f, A_log, Dvec, brY[br]);
    }

    // --- out = (y1 + y2) * silu(xp) + skip ---
    int n = MROWS * DIM;
    combine_kernel<<<(n + 255) / 256, 256, 0, stream>>>(y1, y2, xp_f, x, out, n);
}
